// EnhancedTFNLayer_33243046871119
// MI455X (gfx1250) — matrix operand order repaired and checked by reading, not compiled, not run on hardware
//
#include <hip/hip_runtime.h>

#define BB 8
#define NN 4096
#define DD 512
#define MM 512

typedef __bf16 v16bf __attribute__((ext_vector_type(16)));
typedef __bf16 v2bf  __attribute__((ext_vector_type(2)));
typedef float  v8f   __attribute__((ext_vector_type(8)));

union ABReg { uint4 q[2]; v16bf v; };

__device__ __forceinline__ unsigned short f2bf(float f) {
  unsigned int u = __float_as_uint(f);
  u += 0x7fffu + ((u >> 16) & 1u);        // round-to-nearest-even
  return (unsigned short)(u >> 16);
}

__device__ __forceinline__ unsigned int pack2bf(float a, float b) {
#if __has_builtin(__builtin_amdgcn_cvt_pk_bf16_f32)
  v2bf p = __builtin_amdgcn_cvt_pk_bf16_f32(a, b);
  return __builtin_bit_cast(unsigned int, p);
#else
  return (unsigned int)f2bf(a) | ((unsigned int)f2bf(b) << 16);
#endif
}

__device__ __forceinline__ float fast_tanh(float x) {
  x = fminf(fmaxf(x, -15.0f), 15.0f);
  float e = __expf(2.0f * x);
  return (e - 1.0f) / (e + 1.0f);
}

__device__ __forceinline__ v8f wmma_bf16(const ABReg& a, const ABReg& b, v8f c) {
  return __builtin_amdgcn_wmma_f32_16x16x32_bf16(false, a.v, false, b.v, (short)0, c,
                                                 false, false);
}

// Dual reduction (sum a, sum b) across a 256-thread block: wave32 shuffles +
// one LDS combine (single barrier).
__device__ __forceinline__ float2 block_red2(float a, float b, float* rbuf, int t) {
  #pragma unroll
  for (int m = 16; m > 0; m >>= 1) {
    a += __shfl_xor(a, m, 32);
    b += __shfl_xor(b, m, 32);
  }
  if ((t & 31) == 0) { rbuf[t >> 5] = a; rbuf[8 + (t >> 5)] = b; }
  __syncthreads();
  float sa = 0.f, sb = 0.f;
  #pragma unroll
  for (int i = 0; i < 8; ++i) { sa += rbuf[i]; sb += rbuf[8 + i]; }
  return make_float2(sa, sb);
}

// ---------------------------------------------------------------------------
// Transpose x [B,N,D] f32 -> xT [B,D,N] bf16 (B operand for projector GEMM)
// ---------------------------------------------------------------------------
__global__ __launch_bounds__(256)
void k_xpose(const float* __restrict__ x, unsigned short* __restrict__ xT) {
  __shared__ unsigned short sT[32 * 33];
  int bid = blockIdx.x;
  const int d0 = (bid & 15) << 5;  bid >>= 4;
  const int n0 = (bid & 127) << 5; const int b = bid >> 7;
  const int t = threadIdx.x;
  #pragma unroll
  for (int i = 0; i < 4; ++i) {
    int idx = t + (i << 8);
    int nn = idx >> 5, dd = idx & 31;
    sT[dd * 33 + nn] = f2bf(x[((size_t)(b * NN) + n0 + nn) * DD + d0 + dd]);
  }
  __syncthreads();
  #pragma unroll
  for (int i = 0; i < 4; ++i) {
    int idx = t + (i << 8);
    int dd = idx >> 5, nn = idx & 31;
    xT[((size_t)(b * DD) + d0 + dd) * NN + n0 + nn] = sT[dd * 33 + nn];
  }
}

// ---------------------------------------------------------------------------
// Transpose W [K,COL] f32 -> WT [COL,K] bf16 (col-major B operand)
// ---------------------------------------------------------------------------
__global__ __launch_bounds__(256)
void k_wxpose(const float* __restrict__ W, unsigned short* __restrict__ WT) {
  __shared__ unsigned short sT[32 * 33];
  const int c0 = (blockIdx.x & 15) << 5;
  const int k0 = (blockIdx.x >> 4) << 5;
  const int t = threadIdx.x;
  #pragma unroll
  for (int i = 0; i < 4; ++i) {
    int idx = t + (i << 8);
    int kk = idx >> 5, cc = idx & 31;
    sT[cc * 33 + kk] = f2bf(W[(size_t)(k0 + kk) * DD + c0 + cc]);
  }
  __syncthreads();
  #pragma unroll
  for (int i = 0; i < 4; ++i) {
    int idx = t + (i << 8);
    int cc = idx >> 5, kk = idx & 31;
    WT[(size_t)(c0 + cc) * DD + k0 + kk] = sT[cc * 33 + kk];
  }
}

// ---------------------------------------------------------------------------
// Kernel 1: field[b,m,d] = sum_n exp(-(g_m - p_n)^2/(2 sigma^2)) * x[b,n,d]
// Block tile 32M x 128D, K = 4096. RBF A-tile computed into LDS in 64-wide
// stages (packed b32 stores, 1 barrier pair per 4 WMMAs/wave); B fragments
// loaded directly from xT bf16 global. Epilogue writes f32 + bf16 field.
// ---------------------------------------------------------------------------
__global__ __launch_bounds__(256)
void k_projector(const unsigned short* __restrict__ xT, const float* __restrict__ pos,
                 const float* __restrict__ grid, const float* __restrict__ log_sigma,
                 float* __restrict__ field, unsigned short* __restrict__ fieldb) {
  __shared__ __align__(16) unsigned short sKern[32 * 64];   // A: [m][k-stage]
  __shared__ float sG[32];

  const int t    = threadIdx.x;
  const int lane = t & 31;
  const int wave = t >> 5;
  const int wm   = wave >> 2;
  const int wd   = wave & 3;
  const int row  = lane & 15;
  const int hi   = lane >> 4;

  int bid = blockIdx.x;
  const int d0 = (bid & 3) << 7;  bid >>= 2;
  const int m0 = (bid & 15) << 5; const int b = bid >> 4;

  const float sg   = __expf(log_sigma[0]);
  const float ninv = -0.5f / (sg * sg);

  if (t < 32) sG[t] = grid[b * MM + m0 + t];

  const float* pp = &pos[(size_t)b * NN];
  const unsigned short* bc0 = xT + (size_t)(b * DD + d0 + wd * 32 + row) * NN;
  const unsigned short* bc1 = bc0 + (size_t)16 * NN;

  v8f c0 = {}; v8f c1 = {};

  for (int kt = 0; kt < NN / 64; ++kt) {
    const int n0 = kt << 6;
    __syncthreads();
    // RBF tile 32x64: 4 packed pairs / thread; v_exp_f32 co-issues with WMMA
    #pragma unroll
    for (int i = 0; i < 4; ++i) {
      int p  = t + (i << 8);              // pair index 0..1023
      int mi = p >> 5, j2 = (p & 31) << 1;
      float  g = sG[mi];
      float2 pv = *(const float2*)&pp[n0 + j2];
      float da = g - pv.x;
      float db = g - pv.y;
      *(unsigned int*)&sKern[mi * 64 + j2] =
          pack2bf(__expf(da * da * ninv), __expf(db * db * ninv));
    }
    __syncthreads();

    #pragma unroll
    for (int s = 0; s < 2; ++s) {
      const int nn = n0 + (s << 5);
      ABReg a, b0, b1;
      const unsigned short* ap = &sKern[(wm * 16 + row) * 64 + (s << 5) + hi * 16];
      a.q[0]  = *(const uint4*)(ap);
      a.q[1]  = *(const uint4*)(ap + 8);
      b0.q[0] = *(const uint4*)(bc0 + nn + hi * 16);
      b0.q[1] = *(const uint4*)(bc0 + nn + hi * 16 + 8);
      b1.q[0] = *(const uint4*)(bc1 + nn + hi * 16);
      b1.q[1] = *(const uint4*)(bc1 + nn + hi * 16 + 8);
      c0 = wmma_bf16(a, b0, c0);
      c1 = wmma_bf16(a, b1, c1);
    }
  }

  const int mbase = m0 + wm * 16 + hi * 8;
  const int col0  = d0 + wd * 32 + row;
  #pragma unroll
  for (int r = 0; r < 8; ++r) {
    size_t o = ((size_t)(b * MM) + mbase + r) * DD + col0;
    field[o]       = c0[r];
    field[o + 16]  = c1[r];
    fieldb[o]      = f2bf(c0[r]);
    fieldb[o + 16] = f2bf(c1[r]);
  }
}

// ---------------------------------------------------------------------------
// Kernel 2: one diffusion step, fully LDS/barrier-free.
// out = in + dt*(alpha*lap(in) + tanh(in @ W_int + b_int))
// A: bf16 field copy (row-major) via global_load_b128; B: col-major W_int bf16.
// One wave = one 16x32 tile; K = 512.
// ---------------------------------------------------------------------------
__global__ __launch_bounds__(256)
void k_dyn(const float* __restrict__ fin, const unsigned short* __restrict__ finb,
           float* __restrict__ fout, unsigned short* __restrict__ foutb,
           const unsigned short* __restrict__ WT, const float* __restrict__ bint,
           const float* __restrict__ alpha_p) {
  const int t    = threadIdx.x;
  const int lane = t & 31;
  const int row  = lane & 15;
  const int hi   = lane >> 4;

  const int wg   = blockIdx.x * 8 + (t >> 5);   // 0..4095 tiles
  const int col0 = (wg & 15) << 5;              // 16 col-tiles of 32
  const int mt   = wg >> 4;                     // 256 row-tiles of 16 (B*M rows)

  const unsigned short* arow = finb + (size_t)((mt << 4) + row) * DD;
  const unsigned short* bc0  = WT + (size_t)(col0 + row) * DD;
  const unsigned short* bc1  = bc0 + (size_t)16 * DD;

  v8f c0 = {}; v8f c1 = {};
  for (int k0 = 0; k0 < DD; k0 += 32) {
    ABReg a, b0, b1;
    a.q[0]  = *(const uint4*)(arow + k0 + hi * 16);
    a.q[1]  = *(const uint4*)(arow + k0 + hi * 16 + 8);
    b0.q[0] = *(const uint4*)(bc0 + k0 + hi * 16);
    b0.q[1] = *(const uint4*)(bc0 + k0 + hi * 16 + 8);
    b1.q[0] = *(const uint4*)(bc1 + k0 + hi * 16);
    b1.q[1] = *(const uint4*)(bc1 + k0 + hi * 16 + 8);
    c0 = wmma_bf16(a, b0, c0);
    c1 = wmma_bf16(a, b1, c1);
  }

  const float alpha = alpha_p[0];
  const float dtc   = 0.25f;
  #pragma unroll
  for (int r = 0; r < 8; ++r) {
    int mrow = (mt << 4) + hi * 8 + r;          // flattened b*M + m
    int m    = mrow & (MM - 1);
    int ml   = (m == 0)      ? mrow : mrow - 1; // Neumann edges within batch
    int mh   = (m == MM - 1) ? mrow : mrow + 1;
    #pragma unroll
    for (int s = 0; s < 2; ++s) {
      int col = col0 + s * 16 + row;
      float acc = s ? c1[r] : c0[r];
      size_t o  = (size_t)mrow * DD + col;
      float f   = fin[o];
      float fl  = fin[(size_t)ml * DD + col];
      float fh  = fin[(size_t)mh * DD + col];
      float v = f + dtc * (alpha * (fl - 2.0f * f + fh) + fast_tanh(acc + bint[col]));
      fout[o]  = v;
      foutb[o] = f2bf(v);
    }
  }
}

// ---------------------------------------------------------------------------
// Kernel 3: linear-interp sample + residual + LayerNorm1 -> enhanced (f32)
// ---------------------------------------------------------------------------
__global__ __launch_bounds__(256)
void k_sample_ln1(const float* __restrict__ field, const float* __restrict__ x,
                  const float* __restrict__ pos, const float* __restrict__ g,
                  const float* __restrict__ beta, float* __restrict__ enh) {
  __shared__ float rbuf[16];
  const int t = threadIdx.x;
  const int rowid = blockIdx.x;          // b*N + n
  const int b = rowid >> 12;

  const float u = pos[rowid] * (float)(MM - 1);
  int i0 = (int)floorf(u);
  i0 = i0 < 0 ? 0 : (i0 > MM - 2 ? MM - 2 : i0);
  const float w = u - (float)i0;

  const float* f0p = &field[((size_t)(b * MM) + i0) * DD];
  const float* xp  = &x[(size_t)rowid * DD];

  float a0 = f0p[t],       a1 = f0p[DD + t];
  float b0 = f0p[t + 256], b1 = f0p[DD + t + 256];
  float v0 = a0 + w * (a1 - a0) + xp[t];
  float v1 = b0 + w * (b1 - b0) + xp[t + 256];

  float2 sv = block_red2(v0 + v1, v0 * v0 + v1 * v1, rbuf, t);
  float mu = sv.x * (1.0f / DD);
  float rs = rsqrtf(sv.y * (1.0f / DD) - mu * mu + 1e-5f);

  float* op = &enh[(size_t)rowid * DD];
  op[t]       = (v0 - mu) * rs * g[t]       + beta[t];
  op[t + 256] = (v1 - mu) * rs * g[t + 256] + beta[t + 256];
}

// ---------------------------------------------------------------------------
// Kernel 4: y = enhanced @ W_out + b_out + enhanced  (pre-LN2) -> d_out
// A staged via LDS in 64-wide stages (packed float2->b32 conversion);
// B direct from col-major W_out bf16 in global. Tile 32 x 128, K = 512.
// ---------------------------------------------------------------------------
__global__ __launch_bounds__(256)
void k_out_gemm(const float* __restrict__ enh, const unsigned short* __restrict__ WT,
                const float* __restrict__ bout, float* __restrict__ y) {
  __shared__ __align__(16) unsigned short sA[32 * 64];

  const int t    = threadIdx.x;
  const int lane = t & 31;
  const int wave = t >> 5;
  const int wm   = wave >> 2;
  const int wd   = wave & 3;
  const int row  = lane & 15;
  const int hi   = lane >> 4;

  const int d0   = (blockIdx.x & 3) << 7;
  const int row0 = (blockIdx.x >> 2) << 5;

  const unsigned short* bc0 = WT + (size_t)(d0 + wd * 32 + row) * DD;
  const unsigned short* bc1 = bc0 + (size_t)16 * DD;

  v8f c0 = {}; v8f c1 = {};

  for (int kt = 0; kt < DD / 64; ++kt) {
    const int k0 = kt << 6;
    __syncthreads();
    #pragma unroll
    for (int i = 0; i < 4; ++i) {
      int p  = t + (i << 8);               // 1024 pairs
      int ir = p >> 5, kp = (p & 31) << 1;
      float2 v = *(const float2*)&enh[(size_t)(row0 + ir) * DD + k0 + kp];
      *(unsigned int*)&sA[ir * 64 + kp] = pack2bf(v.x, v.y);
    }
    __syncthreads();

    #pragma unroll
    for (int s = 0; s < 2; ++s) {
      const int ks = k0 + (s << 5);
      ABReg a, b0, b1;
      const unsigned short* ap = &sA[(wm * 16 + row) * 64 + (s << 5) + hi * 16];
      a.q[0]  = *(const uint4*)(ap);
      a.q[1]  = *(const uint4*)(ap + 8);
      b0.q[0] = *(const uint4*)(bc0 + ks + hi * 16);
      b0.q[1] = *(const uint4*)(bc0 + ks + hi * 16 + 8);
      b1.q[0] = *(const uint4*)(bc1 + ks + hi * 16);
      b1.q[1] = *(const uint4*)(bc1 + ks + hi * 16 + 8);
      c0 = wmma_bf16(a, b0, c0);
      c1 = wmma_bf16(a, b1, c1);
    }
  }

  const int rbase = row0 + wm * 16 + hi * 8;
  const int col0  = d0 + wd * 32 + row;
  #pragma unroll
  for (int r = 0; r < 8; ++r) {
    size_t o0 = (size_t)(rbase + r) * DD + col0;
    y[o0]      = c0[r] + bout[col0]      + enh[o0];
    y[o0 + 16] = c1[r] + bout[col0 + 16] + enh[o0 + 16];
  }
}

// ---------------------------------------------------------------------------
// Kernel 5: in-place LayerNorm2 on d_out rows.
// ---------------------------------------------------------------------------
__global__ __launch_bounds__(256)
void k_ln2(float* __restrict__ y, const float* __restrict__ g,
           const float* __restrict__ beta) {
  __shared__ float rbuf[16];
  const int t = threadIdx.x;
  float* yp = &y[(size_t)blockIdx.x * DD];
  float v0 = yp[t], v1 = yp[t + 256];

  float2 sv = block_red2(v0 + v1, v0 * v0 + v1 * v1, rbuf, t);
  float mu = sv.x * (1.0f / DD);
  float rs = rsqrtf(sv.y * (1.0f / DD) - mu * mu + 1e-5f);

  yp[t]       = (v0 - mu) * rs * g[t]       + beta[t];
  yp[t + 256] = (v1 - mu) * rs * g[t + 256] + beta[t + 256];
}

// ---------------------------------------------------------------------------
extern "C" void kernel_launch(void* const* d_in, const int* in_sizes, int n_in,
                              void* d_out, int out_size, void* d_ws, size_t ws_size,
                              hipStream_t stream) {
  const float* x         = (const float*)d_in[0];
  const float* pos       = (const float*)d_in[1];
  const float* grid      = (const float*)d_in[2];
  const float* log_sigma = (const float*)d_in[3];
  const float* alpha     = (const float*)d_in[4];
  const float* W_int     = (const float*)d_in[5];
  const float* b_int     = (const float*)d_in[6];
  const float* ln1_g     = (const float*)d_in[7];
  const float* ln1_b     = (const float*)d_in[8];
  const float* W_out     = (const float*)d_in[9];
  const float* b_out     = (const float*)d_in[10];
  const float* ln2_g     = (const float*)d_in[11];
  const float* ln2_b     = (const float*)d_in[12];
  float* out = (float*)d_out;

  char* ws = (char*)d_ws;
  // region0: xT_bf16 (33.5 MB, used only by projector) aliased with enhanced f32
  unsigned short* xTb = (unsigned short*)ws;
  float* enh          = (float*)ws;                          // 67,108,864 B
  float* fA           = (float*)(ws + 67108864);             //  8,388,608 B
  float* fB           = (float*)(ws + 75497472);             //  8,388,608 B
  unsigned short* fAb = (unsigned short*)(ws + 83886080);    //  4,194,304 B
  unsigned short* fBb = (unsigned short*)(ws + 88080384);    //  4,194,304 B
  unsigned short* WiT = (unsigned short*)(ws + 92274688);    //    524,288 B
  unsigned short* WoT = (unsigned short*)(ws + 92798976);    //    524,288 B

  k_wxpose    <<<256,   256, 0, stream>>>(W_int, WiT);
  k_wxpose    <<<256,   256, 0, stream>>>(W_out, WoT);
  k_xpose     <<<16384, 256, 0, stream>>>(x, xTb);
  k_projector <<<512,   256, 0, stream>>>(xTb, pos, grid, log_sigma, fA, fAb);
  k_dyn       <<<512,   256, 0, stream>>>(fA, fAb, fB, fBb, WiT, b_int, alpha);
  k_dyn       <<<512,   256, 0, stream>>>(fB, fBb, fA, fAb, WiT, b_int, alpha);
  k_dyn       <<<512,   256, 0, stream>>>(fA, fAb, fB, fBb, WiT, b_int, alpha);
  k_dyn       <<<512,   256, 0, stream>>>(fB, fBb, fA, fAb, WiT, b_int, alpha);
  k_sample_ln1<<<32768, 256, 0, stream>>>(fA, x, pos, ln1_g, ln1_b, enh);
  k_out_gemm  <<<4096,  256, 0, stream>>>(enh, WoT, b_out, out);
  k_ln2       <<<32768, 256, 0, stream>>>(out, ln2_g, ln2_b);
}
